// HybridTransformer_65481071410518
// MI455X (gfx1250) — matrix operand order repaired and checked by reading, not compiled, not run on hardware
//
#include <hip/hip_runtime.h>
#include <hip/hip_bf16.h>

typedef _Float16 half_t;
typedef __attribute__((ext_vector_type(16))) _Float16 v16h;
typedef __attribute__((ext_vector_type(8)))  _Float16 v8h;
typedef __attribute__((ext_vector_type(8)))  float    v8f;

#define D_DIM 512
#define F_DIM 2048
#define S_DIM 2048
#define B_DIM 8
#define L_DIM 4
#define BS    (B_DIM * S_DIM)

// ---------------------------------------------------------------------------
// helpers
// ---------------------------------------------------------------------------
__device__ __forceinline__ v16h load_a_frag(const half_t* base, int kbA) {
    // A fragment (16x32 f16): lanes 0-15 rows, halves[0..7]=K[kbA..kbA+8),
    // halves[8..15]=K[kbA+8..kbA+16)
    v8h lo = *(const v8h*)(base + kbA);
    v8h hi = *(const v8h*)(base + kbA + 8);
    return __builtin_shufflevector(lo, hi, 0,1,2,3,4,5,6,7,8,9,10,11,12,13,14,15);
}

__device__ __forceinline__ v8f wmma_f16(v16h a, v16h b, v8f c) {
    return __builtin_amdgcn_wmma_f32_16x16x32_f16(false, a, false, b,
                                                  (short)0, c, false, false);
}

// ---------------------------------------------------------------------------
// weight transpose + convert:  out[n*K + k] = (f16) in[k*N + n]
// ---------------------------------------------------------------------------
__global__ __launch_bounds__(256)
void transpose_f32_to_f16(const float* __restrict__ in, half_t* __restrict__ out,
                          int K, int N) {
    int idx = blockIdx.x * 256 + threadIdx.x;
    if (idx < K * N) {
        int n = idx / K;
        int k = idx - n * K;
        out[idx] = (half_t)in[(size_t)k * N + n];
    }
}

// ---------------------------------------------------------------------------
// embedding + sinusoidal PE -> x (f32) and xh (f16)
// ---------------------------------------------------------------------------
__global__ __launch_bounds__(256)
void embed_pe_kernel(const int* __restrict__ tokens, const float* __restrict__ emb,
                     float* __restrict__ x, half_t* __restrict__ xh) {
    int row = blockIdx.x;                  // b*S + s
    int s   = row & (S_DIM - 1);
    int tok = tokens[row];
    const float* er = emb + (size_t)tok * D_DIM;
    int t = threadIdx.x;
    #pragma unroll
    for (int off = 0; off < D_DIM; off += 256) {
        int d = t + off;
        // div = exp(-(ln 10000)/D * (d & ~1))
        float div = __expf((float)(d & ~1) * -0.017988946039015984f);
        float ang = (float)s * div;
        float pe  = (d & 1) ? __cosf(ang) : __sinf(ang);
        float val = er[d] + pe;
        x [(size_t)row * D_DIM + d] = val;
        xh[(size_t)row * D_DIM + d] = (half_t)val;
    }
}

// ---------------------------------------------------------------------------
// Generic WMMA GEMM:  C[M,N] = A[M,K] @ W[K,N] + bias, W stored transposed [N,K].
// MODE 0: out f16 row-major.  MODE 1: out f16 transposed [B,N,S] (for V).
// MODE 2: out f32 row-major with residual add.
// Block: 256 threads = 8 waves. WG tile 128(M) x 64(N); each wave: 16x64.
// ---------------------------------------------------------------------------
template<int K, int MODE, bool RELU>
__global__ __launch_bounds__(256)
void gemm_wmma(const half_t* __restrict__ A, const half_t* __restrict__ Wt,
               const float* __restrict__ bias, const float* __restrict__ resid,
               half_t* __restrict__ outH, float* __restrict__ outF, int N) {
    const int lane = threadIdx.x & 31;
    const int wave = threadIdx.x >> 5;
    const int lm   = lane & 15;
    const int hi   = lane >> 4;
    const int m0   = blockIdx.x * 128 + wave * 16;
    const int n0   = blockIdx.y * 64;
    const int kbA  = hi ? 16 : 0;
    const int kbB  = hi ? 16 : 0;

    v8f zero = {0.f,0.f,0.f,0.f,0.f,0.f,0.f,0.f};
    v8f acc[4] = {zero, zero, zero, zero};

    const half_t* arow = A + (size_t)(m0 + lm) * K;
    for (int k0 = 0; k0 < K; k0 += 32) {
        v16h a = load_a_frag(arow + k0, kbA);
        #pragma unroll
        for (int j = 0; j < 4; ++j) {
            const half_t* bp = Wt + (size_t)(n0 + j * 16 + lm) * K + k0 + kbB;
            v16h b = *(const v16h*)bp;
            acc[j] = wmma_f16(a, b, acc[j]);
        }
    }

    // C/D layout: lane n = lm, rows m = r + 8*hi
    const int mbase = m0 + hi * 8;
    #pragma unroll
    for (int j = 0; j < 4; ++j) {
        int n = n0 + j * 16 + lm;
        float bv = bias[n];
        #pragma unroll
        for (int r = 0; r < 8; ++r) {
            int m = mbase + r;
            float val = acc[j][r] + bv;
            if (RELU) val = val > 0.f ? val : 0.f;
            if (MODE == 2) {
                outF[(size_t)m * N + n] = val + resid[(size_t)m * N + n];
            } else if (MODE == 1) {
                int bb = m / S_DIM, ss = m & (S_DIM - 1);
                outH[((size_t)bb * N + n) * S_DIM + ss] = (half_t)val;
            } else {
                outH[(size_t)m * N + n] = (half_t)val;
            }
        }
    }
}

// ---------------------------------------------------------------------------
// Flash attention (single head, full width D=512, scale 1/sqrt(D)).
// Block: 256 threads (8 waves), 16 queries/block, 128-key blocks, online softmax.
// qh,kh: [B*S,D] f16;  vt: [B,D,S] f16;  outh: [B*S,D] f16.
// ---------------------------------------------------------------------------
__global__ __launch_bounds__(256)
void flash_attn_kernel(const half_t* __restrict__ qh, const half_t* __restrict__ kh,
                       const half_t* __restrict__ vt, half_t* __restrict__ outh) {
    constexpr int KBLK = 128;
    __shared__ float  sS[16][KBLK];
    __shared__ __align__(16) half_t sP[16][KBLK + 8];
    __shared__ float  sAlpha[16], sM[16], sL[16];

    const int lane = threadIdx.x & 31;
    const int wave = threadIdx.x >> 5;
    const int lm   = lane & 15;
    const int hi   = lane >> 4;
    const int kbA  = hi ? 16 : 0;
    const int kbB  = hi ? 16 : 0;
    const int b    = blockIdx.x / (S_DIM / 16);
    const int q0   = (blockIdx.x % (S_DIM / 16)) * 16;

    if (threadIdx.x < 16) { sM[threadIdx.x] = -3.0e38f; sL[threadIdx.x] = 0.f; }

    v8f zero = {0.f,0.f,0.f,0.f,0.f,0.f,0.f,0.f};
    v8f accO[4] = {zero, zero, zero, zero};

    const half_t* qrow  = qh + (size_t)(b * S_DIM + q0 + lm) * D_DIM;
    const half_t* vbase = vt + (size_t)b * D_DIM * S_DIM;
    const float scale = 0.04419417382415922f;   // 1/sqrt(512)

    __syncthreads();

    for (int kb = 0; kb < S_DIM; kb += KBLK) {
        // ---- scores: wave w owns keys [kb + 16w, kb + 16w + 16) ----
        v8f accS = zero;
        const half_t* krow = kh + (size_t)(b * S_DIM + kb + wave * 16 + lm) * D_DIM;
        #pragma unroll
        for (int k0 = 0; k0 < D_DIM; k0 += 32) {
            v16h a = load_a_frag(qrow + k0, kbA);
            v16h bf = *(const v16h*)(krow + k0 + kbB);   // K natural: contiguous d
            accS = wmma_f16(a, bf, accS);
        }
        #pragma unroll
        for (int r = 0; r < 8; ++r)
            sS[r + hi * 8][wave * 16 + lm] = accS[r] * scale;
        __syncthreads();

        // ---- online softmax for this key block (one thread per query row) ----
        if (threadIdx.x < 16) {
            int rq = threadIdx.x;
            float mold = sM[rq], mnew = mold;
            for (int j = 0; j < KBLK; ++j) mnew = fmaxf(mnew, sS[rq][j]);
            float alpha = __expf(mold - mnew);
            float sum = 0.f;
            for (int j = 0; j < KBLK; ++j) {
                float p = __expf(sS[rq][j] - mnew);
                sP[rq][j] = (half_t)p;
                sum += p;
            }
            sM[rq] = mnew;
            sL[rq] = sL[rq] * alpha + sum;
            sAlpha[rq] = alpha;
        }
        __syncthreads();

        // ---- rescale accumulators, then accO += P @ V (wave owns 64 d-cols) ----
        float al[8];
        #pragma unroll
        for (int r = 0; r < 8; ++r) al[r] = sAlpha[r + hi * 8];
        #pragma unroll
        for (int j = 0; j < 4; ++j)
            #pragma unroll
            for (int r = 0; r < 8; ++r) accO[j][r] *= al[r];

        #pragma unroll
        for (int kc = 0; kc < KBLK; kc += 32) {
            v16h aP = load_a_frag(&sP[lm][kc], kbA);
            #pragma unroll
            for (int j = 0; j < 4; ++j) {
                int n = wave * 64 + j * 16 + lm;
                v16h bV = *(const v16h*)(vbase + (size_t)n * S_DIM + kb + kc + kbB);
                accO[j] = wmma_f16(aP, bV, accO[j]);
            }
        }
        __syncthreads();
    }

    // ---- normalize by running sum, store f16 ----
    float invl[8];
    #pragma unroll
    for (int r = 0; r < 8; ++r) invl[r] = 1.f / sL[r + hi * 8];
    #pragma unroll
    for (int j = 0; j < 4; ++j) {
        int n = wave * 64 + j * 16 + lm;
        #pragma unroll
        for (int r = 0; r < 8; ++r) {
            int q = q0 + r + hi * 8;
            outh[(size_t)(b * S_DIM + q) * D_DIM + n] = (half_t)(accO[j][r] * invl[r]);
        }
    }
}

// ---------------------------------------------------------------------------
// LayerNorm over D=512: y -> x (f32) and xh (f16). One block per row.
// ---------------------------------------------------------------------------
__global__ __launch_bounds__(256)
void layernorm_kernel(const float* __restrict__ y, const float* __restrict__ g,
                      const float* __restrict__ bta, float* __restrict__ xo,
                      half_t* __restrict__ xh) {
    __shared__ float red[256];
    const size_t row = blockIdx.x;
    const float* yr = y + row * D_DIM;
    int t = threadIdx.x;
    float v0 = yr[t], v1 = yr[t + 256];
    red[t] = v0 + v1; __syncthreads();
    for (int s = 128; s > 0; s >>= 1) { if (t < s) red[t] += red[t + s]; __syncthreads(); }
    float mu = red[0] * (1.f / D_DIM);
    __syncthreads();
    float d0 = v0 - mu, d1 = v1 - mu;
    red[t] = d0 * d0 + d1 * d1; __syncthreads();
    for (int s = 128; s > 0; s >>= 1) { if (t < s) red[t] += red[t + s]; __syncthreads(); }
    float inv = rsqrtf(red[0] * (1.f / D_DIM) + 1e-5f);
    float o0 = d0 * inv * g[t]       + bta[t];
    float o1 = d1 * inv * g[t + 256] + bta[t + 256];
    xo[row * D_DIM + t]       = o0;
    xo[row * D_DIM + t + 256] = o1;
    xh[row * D_DIM + t]       = (half_t)o0;
    xh[row * D_DIM + t + 256] = (half_t)o1;
}

// ---------------------------------------------------------------------------
// mean-pool over S then dot with clf_w: out[b] = mean_s(x[b,s,:].w) + clf_b
// ---------------------------------------------------------------------------
__global__ __launch_bounds__(256)
void classify_kernel(const float* __restrict__ x, const float* __restrict__ w,
                     const float* __restrict__ bclf, float* __restrict__ out) {
    __shared__ float red[256];
    int b = blockIdx.x;
    const float* xb = x + (size_t)b * S_DIM * D_DIM;
    float sum = 0.f;
    for (int idx = threadIdx.x; idx < S_DIM * D_DIM; idx += 256)
        sum += xb[idx] * w[idx & (D_DIM - 1)];
    red[threadIdx.x] = sum; __syncthreads();
    for (int s = 128; s > 0; s >>= 1) {
        if (threadIdx.x < s) red[threadIdx.x] += red[threadIdx.x + s];
        __syncthreads();
    }
    if (threadIdx.x == 0) out[b] = red[0] * (1.f / S_DIM) + bclf[0];
}

// ---------------------------------------------------------------------------
// host launcher
// ---------------------------------------------------------------------------
extern "C" void kernel_launch(void* const* d_in, const int* in_sizes, int n_in,
                              void* d_out, int out_size, void* d_ws, size_t ws_size,
                              hipStream_t stream) {
    const int*   tokens = (const int*)  d_in[0];
    const float* emb    = (const float*)d_in[1];
    const float* Wq = (const float*)d_in[2],  *bq = (const float*)d_in[3];
    const float* Wk = (const float*)d_in[4],  *bk = (const float*)d_in[5];
    const float* Wv = (const float*)d_in[6],  *bv = (const float*)d_in[7];
    const float* Wo = (const float*)d_in[8],  *bo = (const float*)d_in[9];
    const float* W1 = (const float*)d_in[10], *b1 = (const float*)d_in[11];
    const float* W2 = (const float*)d_in[12], *b2 = (const float*)d_in[13];
    const float* ln1g = (const float*)d_in[14], *ln1b = (const float*)d_in[15];
    const float* ln2g = (const float*)d_in[16], *ln2b = (const float*)d_in[17];
    const float* clfw = (const float*)d_in[18], *clfb = (const float*)d_in[19];
    float* outp = (float*)d_out;

    // workspace carve-up
    char* ws = (char*)d_ws;
    size_t off = 0;
    auto carve = [&](size_t bytes) {
        void* p = ws + off;
        off += (bytes + 255) & ~(size_t)255;
        return p;
    };
    const size_t WLAYER = 4 * (size_t)D_DIM * D_DIM + 2 * (size_t)D_DIM * F_DIM;
    half_t* wT    = (half_t*)carve(L_DIM * WLAYER * sizeof(half_t));
    half_t* xh    = (half_t*)carve((size_t)BS * D_DIM * sizeof(half_t));
    half_t* qh    = (half_t*)carve((size_t)BS * D_DIM * sizeof(half_t));
    half_t* kh    = (half_t*)carve((size_t)BS * D_DIM * sizeof(half_t));
    half_t* vt    = (half_t*)carve((size_t)BS * D_DIM * sizeof(half_t));
    half_t* atth  = (half_t*)carve((size_t)BS * D_DIM * sizeof(half_t));
    half_t* hh    = (half_t*)carve((size_t)BS * F_DIM * sizeof(half_t));
    float*  x     = (float*) carve((size_t)BS * D_DIM * sizeof(float));
    float*  y     = (float*) carve((size_t)BS * D_DIM * sizeof(float));

    // 1) transpose+convert all weights to f16 [N,K]
    for (int l = 0; l < L_DIM; ++l) {
        half_t* lw = wT + l * WLAYER;
        const int gDD = (D_DIM * D_DIM + 255) / 256;
        const int gDF = (D_DIM * F_DIM + 255) / 256;
        transpose_f32_to_f16<<<gDD, 256, 0, stream>>>(Wq + (size_t)l*D_DIM*D_DIM, lw,                         D_DIM, D_DIM);
        transpose_f32_to_f16<<<gDD, 256, 0, stream>>>(Wk + (size_t)l*D_DIM*D_DIM, lw + 1*(size_t)D_DIM*D_DIM, D_DIM, D_DIM);
        transpose_f32_to_f16<<<gDD, 256, 0, stream>>>(Wv + (size_t)l*D_DIM*D_DIM, lw + 2*(size_t)D_DIM*D_DIM, D_DIM, D_DIM);
        transpose_f32_to_f16<<<gDD, 256, 0, stream>>>(Wo + (size_t)l*D_DIM*D_DIM, lw + 3*(size_t)D_DIM*D_DIM, D_DIM, D_DIM);
        transpose_f32_to_f16<<<gDF, 256, 0, stream>>>(W1 + (size_t)l*D_DIM*F_DIM, lw + 4*(size_t)D_DIM*D_DIM, D_DIM, F_DIM);
        transpose_f32_to_f16<<<gDF, 256, 0, stream>>>(W2 + (size_t)l*F_DIM*D_DIM, lw + 4*(size_t)D_DIM*D_DIM + (size_t)D_DIM*F_DIM, F_DIM, D_DIM);
    }

    // 2) embedding + positional encoding
    embed_pe_kernel<<<BS, 256, 0, stream>>>(tokens, emb, x, xh);

    dim3 gD(BS / 128, D_DIM / 64);   // 128 x 8
    dim3 gF(BS / 128, F_DIM / 64);   // 128 x 32

    // 3) layers
    for (int l = 0; l < L_DIM; ++l) {
        half_t* lw = wT + l * WLAYER;
        half_t* WqT = lw;
        half_t* WkT = lw + 1 * (size_t)D_DIM * D_DIM;
        half_t* WvT = lw + 2 * (size_t)D_DIM * D_DIM;
        half_t* WoT = lw + 3 * (size_t)D_DIM * D_DIM;
        half_t* W1T = lw + 4 * (size_t)D_DIM * D_DIM;
        half_t* W2T = W1T + (size_t)D_DIM * F_DIM;

        gemm_wmma<D_DIM, 0, false><<<gD, 256, 0, stream>>>(xh, WqT, bq + l*D_DIM, nullptr, qh, nullptr, D_DIM);
        gemm_wmma<D_DIM, 0, false><<<gD, 256, 0, stream>>>(xh, WkT, bk + l*D_DIM, nullptr, kh, nullptr, D_DIM);
        gemm_wmma<D_DIM, 1, false><<<gD, 256, 0, stream>>>(xh, WvT, bv + l*D_DIM, nullptr, vt, nullptr, D_DIM);

        flash_attn_kernel<<<B_DIM * (S_DIM / 16), 256, 0, stream>>>(qh, kh, vt, atth);

        gemm_wmma<D_DIM, 2, false><<<gD, 256, 0, stream>>>(atth, WoT, bo + l*D_DIM, x, nullptr, y, D_DIM);
        layernorm_kernel<<<BS, 256, 0, stream>>>(y, ln1g + l*D_DIM, ln1b + l*D_DIM, x, xh);

        gemm_wmma<D_DIM, 0, true ><<<gF, 256, 0, stream>>>(xh, W1T, b1 + l*F_DIM, nullptr, hh, nullptr, F_DIM);
        gemm_wmma<F_DIM, 2, false><<<gD, 256, 0, stream>>>(hh, W2T, b2 + l*D_DIM, x, nullptr, y, D_DIM);
        layernorm_kernel<<<BS, 256, 0, stream>>>(y, ln2g + l*D_DIM, ln2b + l*D_DIM, x, xh);
    }

    // 4) mean pool + classifier
    classify_kernel<<<B_DIM, 256, 0, stream>>>(x, clfw, clfb, outp);
}
